// ComplexEndToEndModel_43662637532047
// MI455X (gfx1250) — matrix operand order repaired and checked by reading, not compiled, not run on hardware
//
#include <hip/hip_runtime.h>
#include <math.h>

// ---------------- WMMA types / helpers (gfx1250, wave32) ----------------
typedef __attribute__((ext_vector_type(16))) _Float16 v16h;
typedef __attribute__((ext_vector_type(8)))  float    v8f;

struct alignas(16) U4 { unsigned int a, b, c, d; };
union FragH { v16h v; U4 q[2]; unsigned int u[8]; };

__device__ inline v8f wmma_f16(v16h a, v16h b, v8f c) {
  return __builtin_amdgcn_wmma_f32_16x16x32_f16(
      /*neg_a=*/false, a, /*neg_b=*/false, b,
      /*c_mod=*/(short)0, c, /*reuse_a=*/false, /*reuse_b=*/false);
}

// A fragment: 16x32 f16, rows M = wm*16.. ; LDS layout As[64][32] row-major.
__device__ inline v16h load_frag_a(const _Float16* As, int lane, int wm) {
  FragH f;
  int row = wm * 16 + (lane & 15);
  int kb  = (lane >> 4) * 8;
  const _Float16* p = As + row * 32;
  f.q[0] = *(const U4*)(p + kb);        // K = kb .. kb+7
  f.q[1] = *(const U4*)(p + kb + 16);   // K = kb+16 .. kb+23
  return f.v;
}
// B fragment: 32x16 f16, cols N = nbase.. ; LDS layout Bs[64][32] = [n][k].
__device__ inline v16h load_frag_b(const _Float16* Bs, int lane, int nbase) {
  FragH f;
  int row = nbase + (lane & 15);
  int kb  = (lane >> 4) * 8;
  const _Float16* p = Bs + row * 32;
  f.q[0] = *(const U4*)(p + kb);
  f.q[1] = *(const U4*)(p + kb + 16);
  return f.v;
}
__device__ inline v16h neg_frag(v16h a) {
  FragH f; f.v = a;
#pragma unroll
  for (int i = 0; i < 8; ++i) f.u[i] ^= 0x80008000u;
  return f.v;
}
#define ZERO8 {0.f,0.f,0.f,0.f,0.f,0.f,0.f,0.f}

// ---------------- complex conv as implicit-im2col WMMA GEMM ----------------
// Y[b][co][ho][wo] = sum_{ci,kh,kw} W[co][ci][kh][kw] * X[b][ci][hi][wi]  (complex)
__global__ __launch_bounds__(256)
void k_cconv_wmma(const float* __restrict__ Xr, const float* __restrict__ Xi,
                  const float* __restrict__ Wr, const float* __restrict__ Wi,
                  const float* __restrict__ BiasR, const float* __restrict__ BiasI,
                  float* __restrict__ Yr, float* __restrict__ Yi,
                  int NB, int Cin, int Hin, int Win,
                  int Cout, int Ho, int Wo, int KS, int stride, int pad) {
  __shared__ alignas(16) _Float16 Asr[64 * 32];
  __shared__ alignas(16) _Float16 Asi[64 * 32];
  __shared__ alignas(16) _Float16 Bsr[64 * 32];
  __shared__ alignas(16) _Float16 Bsi[64 * 32];
  const int tid  = threadIdx.x;
  const int lane = tid & 31, wave = tid >> 5;
  const int wm = wave & 3, wn = wave >> 2;
  const int tileN = blockIdx.x * 64, tileM = blockIdx.y * 64;
  const int HWo = Ho * Wo, HWi = Hin * Win;
  const int Ntot = NB * HWo, Ktot = Cin * KS * KS, KK = KS * KS;
  v8f accr0 = ZERO8, accr1 = ZERO8, acci0 = ZERO8, acci1 = ZERO8;

  for (int k0 = 0; k0 < Ktot; k0 += 32) {
    // stage A (weights, [Cout][Cin*KS*KS] row-major == native layout)
    for (int idx = tid; idx < 64 * 32; idx += 256) {
      int m = idx >> 5, k = idx & 31;
      int gm = tileM + m, gk = k0 + k;
      float ar = 0.f, ai = 0.f;
      if (gm < Cout && gk < Ktot) {
        size_t o = (size_t)gm * Ktot + gk;
        ar = Wr[o]; ai = Wi[o];
        if (gk + 32 < Ktot) { __builtin_prefetch(&Wr[o + 32], 0, 0); }
      }
      Asr[idx] = (_Float16)ar; Asi[idx] = (_Float16)ai;
    }
    // stage B (implicit im2col), transposed into [n][k]
    for (int idx = tid; idx < 64 * 32; idx += 256) {
      int n = idx >> 5, k = idx & 31;
      int gn = tileN + n, gk = k0 + k;
      float br = 0.f, bi = 0.f;
      if (gn < Ntot && gk < Ktot) {
        int ci = gk / KK, rem = gk - ci * KK;
        int kh = rem / KS, kw = rem - kh * KS;
        int b  = gn / HWo, r2 = gn - b * HWo;
        int ho = r2 / Wo,  wo = r2 - ho * Wo;
        int hi = ho * stride - pad + kh, wi = wo * stride - pad + kw;
        if ((unsigned)hi < (unsigned)Hin && (unsigned)wi < (unsigned)Win) {
          size_t off = ((size_t)(b * Cin + ci)) * HWi + (size_t)hi * Win + wi;
          br = Xr[off]; bi = Xi[off];
        }
      }
      Bsr[idx] = (_Float16)br; Bsi[idx] = (_Float16)bi;
    }
    __syncthreads();
    v16h ar = load_frag_a(Asr, lane, wm);
    v16h ai = load_frag_a(Asi, lane, wm);
    v16h an = neg_frag(ai);
    {
      v16h br = load_frag_b(Bsr, lane, wn * 32);
      v16h bi = load_frag_b(Bsi, lane, wn * 32);
      accr0 = wmma_f16(ar, br, accr0);
      accr0 = wmma_f16(an, bi, accr0);
      acci0 = wmma_f16(ar, bi, acci0);
      acci0 = wmma_f16(ai, br, acci0);
    }
    {
      v16h br = load_frag_b(Bsr, lane, wn * 32 + 16);
      v16h bi = load_frag_b(Bsi, lane, wn * 32 + 16);
      accr1 = wmma_f16(ar, br, accr1);
      accr1 = wmma_f16(an, bi, accr1);
      acci1 = wmma_f16(ar, bi, acci1);
      acci1 = wmma_f16(ai, br, acci1);
    }
    __syncthreads();
  }
  // epilogue: C/D layout -> m = v + 8*(lane>>4), n = lane&15
#pragma unroll
  for (int sub = 0; sub < 2; ++sub) {
    v8f ar = sub ? accr1 : accr0;
    v8f ai = sub ? acci1 : acci0;
    int gn = tileN + wn * 32 + sub * 16 + (lane & 15);
#pragma unroll
    for (int v = 0; v < 8; ++v) {
      int gm = tileM + wm * 16 + v + ((lane >> 4) << 3);
      if (gm < Cout && gn < Ntot) {
        int b = gn / HWo, r2 = gn - b * HWo;
        size_t off = ((size_t)(b * Cout + gm)) * HWo + r2;
        Yr[off] = ar[v] + BiasR[gm];
        Yi[off] = ai[v] + BiasI[gm];
      }
    }
  }
}

// ---------------- attention scores: S = scale*(qr^T kr + qi^T ki) ----------------
// q,k planar [C][N] (per batch, pointers pre-offset). S is [N][N].
__global__ __launch_bounds__(256)
void k_attn_scores(const float* __restrict__ qr, const float* __restrict__ qi,
                   const float* __restrict__ kr, const float* __restrict__ ki,
                   float* __restrict__ S, int C, int N, float scale) {
  __shared__ alignas(16) _Float16 Aqr[64 * 32];
  __shared__ alignas(16) _Float16 Aqi[64 * 32];
  __shared__ alignas(16) _Float16 Bkr[64 * 32];
  __shared__ alignas(16) _Float16 Bki[64 * 32];
  const int tid = threadIdx.x, lane = tid & 31, wave = tid >> 5;
  const int wm = wave & 3, wn = wave >> 2;
  const int tileN = blockIdx.x * 64, tileM = blockIdx.y * 64;
  v8f acc0 = ZERO8, acc1 = ZERO8;

  for (int k0 = 0; k0 < C; k0 += 32) {
    for (int idx = tid; idx < 64 * 32; idx += 256) {
      int m = idx >> 5, k = idx & 31;
      int gm = tileM + m, gk = k0 + k;
      float a0 = 0.f, a1 = 0.f;
      if (gm < N && gk < C) { size_t o = (size_t)gk * N + gm; a0 = qr[o]; a1 = qi[o]; }
      Aqr[idx] = (_Float16)a0; Aqi[idx] = (_Float16)a1;
    }
    for (int idx = tid; idx < 64 * 32; idx += 256) {
      int n = idx >> 5, k = idx & 31;
      int gn = tileN + n, gk = k0 + k;
      float b0 = 0.f, b1 = 0.f;
      if (gn < N && gk < C) { size_t o = (size_t)gk * N + gn; b0 = kr[o]; b1 = ki[o]; }
      Bkr[idx] = (_Float16)b0; Bki[idx] = (_Float16)b1;
    }
    __syncthreads();
    v16h ar = load_frag_a(Aqr, lane, wm);
    v16h ai = load_frag_a(Aqi, lane, wm);
    {
      v16h br = load_frag_b(Bkr, lane, wn * 32);
      v16h bi = load_frag_b(Bki, lane, wn * 32);
      acc0 = wmma_f16(ar, br, acc0);
      acc0 = wmma_f16(ai, bi, acc0);
    }
    {
      v16h br = load_frag_b(Bkr, lane, wn * 32 + 16);
      v16h bi = load_frag_b(Bki, lane, wn * 32 + 16);
      acc1 = wmma_f16(ar, br, acc1);
      acc1 = wmma_f16(ai, bi, acc1);
    }
    __syncthreads();
  }
#pragma unroll
  for (int sub = 0; sub < 2; ++sub) {
    v8f a = sub ? acc1 : acc0;
    int gn = tileN + wn * 32 + sub * 16 + (lane & 15);
#pragma unroll
    for (int v = 0; v < 8; ++v) {
      int gm = tileM + wm * 16 + v + ((lane >> 4) << 3);
      if (gm < N && gn < N) S[(size_t)gm * N + gn] = a[v] * scale;
    }
  }
}

// ---------------- attention apply: OUT[c][n] = X + gamma * sum_m V[c][m]*AW[n][m] ----------------
__global__ __launch_bounds__(256)
void k_attn_out(const float* __restrict__ vr, const float* __restrict__ vi,
                const float* __restrict__ AW,
                const float* __restrict__ xr, const float* __restrict__ xi,
                const float* __restrict__ gammaP,
                float* __restrict__ Or, float* __restrict__ Oi, int C, int N) {
  __shared__ alignas(16) _Float16 Avr[64 * 32];
  __shared__ alignas(16) _Float16 Avi[64 * 32];
  __shared__ alignas(16) _Float16 Bw[64 * 32];
  const int tid = threadIdx.x, lane = tid & 31, wave = tid >> 5;
  const int wm = wave & 3, wn = wave >> 2;
  const int tileN = blockIdx.x * 64, tileM = blockIdx.y * 64;
  v8f ar0 = ZERO8, ar1 = ZERO8, ai0 = ZERO8, ai1 = ZERO8;

  for (int k0 = 0; k0 < N; k0 += 32) {
    for (int idx = tid; idx < 64 * 32; idx += 256) {
      int m = idx >> 5, k = idx & 31;
      int gm = tileM + m, gk = k0 + k;
      float a0 = 0.f, a1 = 0.f;
      if (gm < C && gk < N) { size_t o = (size_t)gm * N + gk; a0 = vr[o]; a1 = vi[o]; }
      Avr[idx] = (_Float16)a0; Avi[idx] = (_Float16)a1;
    }
    for (int idx = tid; idx < 64 * 32; idx += 256) {
      int n = idx >> 5, k = idx & 31;
      int gn = tileN + n, gk = k0 + k;
      float b0 = 0.f;
      if (gn < N && gk < N) b0 = AW[(size_t)gn * N + gk];
      Bw[idx] = (_Float16)b0;
    }
    __syncthreads();
    v16h fr = load_frag_a(Avr, lane, wm);
    v16h fi = load_frag_a(Avi, lane, wm);
    {
      v16h bw = load_frag_b(Bw, lane, wn * 32);
      ar0 = wmma_f16(fr, bw, ar0);
      ai0 = wmma_f16(fi, bw, ai0);
    }
    {
      v16h bw = load_frag_b(Bw, lane, wn * 32 + 16);
      ar1 = wmma_f16(fr, bw, ar1);
      ai1 = wmma_f16(fi, bw, ai1);
    }
    __syncthreads();
  }
  float g = gammaP[0];
#pragma unroll
  for (int sub = 0; sub < 2; ++sub) {
    v8f r = sub ? ar1 : ar0;
    v8f i = sub ? ai1 : ai0;
    int gn = tileN + wn * 32 + sub * 16 + (lane & 15);
#pragma unroll
    for (int v = 0; v < 8; ++v) {
      int gm = tileM + wm * 16 + v + ((lane >> 4) << 3);
      if (gm < C && gn < N) {
        size_t off = (size_t)gm * N + gn;
        Or[off] = xr[off] + g * r[v];
        Oi[off] = xi[off] + g * i[v];
      }
    }
  }
}

// ---------------- small kernels ----------------
__global__ void k_deint(const float* __restrict__ in, float* __restrict__ re,
                        float* __restrict__ im, int n) {
  int i = blockIdx.x * 256 + threadIdx.x;
  if (i < n) { re[i] = in[2 * i]; im[i] = in[2 * i + 1]; }
}

__global__ __launch_bounds__(256)
void k_bn_stats(const float* __restrict__ X, float* __restrict__ mean,
                float* __restrict__ istd, int NB, int C, int HW) {
  int c = blockIdx.x, tid = threadIdx.x;
  int total = NB * HW;
  float s = 0.f, s2 = 0.f;
  for (int i = tid; i < total; i += 256) {
    int b = i / HW, j = i - b * HW;
    float x = X[((size_t)(b * C + c)) * HW + j];
    s += x; s2 += x * x;
  }
  __shared__ float sh[256], sh2[256];
  sh[tid] = s; sh2[tid] = s2; __syncthreads();
  for (int st = 128; st > 0; st >>= 1) {
    if (tid < st) { sh[tid] += sh[tid + st]; sh2[tid] += sh2[tid + st]; }
    __syncthreads();
  }
  if (tid == 0) {
    float m = sh[0] / total;
    float v = sh2[0] / total - m * m;
    mean[c] = m; istd[c] = rsqrtf(v + 1e-5f);
  }
}

__global__ void k_bn_apply(const float* __restrict__ X, float* __restrict__ Y,
                           const float* __restrict__ mean, const float* __restrict__ istd,
                           const float* __restrict__ g, const float* __restrict__ bta,
                           int C, int HW, int total, int relu) {
  int i = blockIdx.x * 256 + threadIdx.x;
  if (i >= total) return;
  int c = (i / HW) % C;
  float y = (X[i] - mean[c]) * istd[c] * g[c] + bta[c];
  if (relu) y = fmaxf(y, 0.f);
  Y[i] = y;
}

__global__ void k_add_relu(const float* __restrict__ a, const float* __restrict__ b,
                           float* __restrict__ o, int n) {
  int i = blockIdx.x * 256 + threadIdx.x;
  if (i < n) o[i] = fmaxf(a[i] + b[i], 0.f);
}

__global__ __launch_bounds__(256)
void k_softmax_rows(float* __restrict__ S, int N) {
  float* row = S + (size_t)blockIdx.x * N;
  int tid = threadIdx.x;
  __shared__ float sh[256];
  float mx = -1e30f;
  for (int i = tid; i < N; i += 256) mx = fmaxf(mx, row[i]);
  sh[tid] = mx; __syncthreads();
  for (int st = 128; st > 0; st >>= 1) {
    if (tid < st) sh[tid] = fmaxf(sh[tid], sh[tid + st]);
    __syncthreads();
  }
  mx = sh[0]; __syncthreads();
  float sum = 0.f;
  for (int i = tid; i < N; i += 256) { float e = expf(row[i] - mx); row[i] = e; sum += e; }
  sh[tid] = sum; __syncthreads();
  for (int st = 128; st > 0; st >>= 1) {
    if (tid < st) sh[tid] += sh[tid + st];
    __syncthreads();
  }
  float inv = 1.f / sh[0];
  for (int i = tid; i < N; i += 256) row[i] *= inv;
}

__global__ void k_concat3(const float* __restrict__ a, const float* __restrict__ b,
                          const float* __restrict__ c, float* __restrict__ out,
                          int N, int total) {
  int i = blockIdx.x * 256 + threadIdx.x;
  if (i >= total) return;
  int n = i % N, t = i / N, cc = t % 384, bb = t / 384;
  const float* src; int cs;
  if (cc < 128)      { src = a; cs = cc; }
  else if (cc < 256) { src = b; cs = cc - 128; }
  else               { src = c; cs = cc - 256; }
  out[i] = src[((size_t)(bb * 128 + cs)) * N + n];
}

__global__ void k_gate_softmax(const float* __restrict__ gr, const float* __restrict__ gi,
                               float* __restrict__ w, int N, int total) {
  int i = blockIdx.x * 256 + threadIdx.x;
  if (i >= total) return;
  int b = i / N, n = i - b * N;
  float m[3];
#pragma unroll
  for (int j = 0; j < 3; ++j) {
    size_t o = ((size_t)(b * 3 + j)) * N + n;
    m[j] = sqrtf(gr[o] * gr[o] + gi[o] * gi[o]);
  }
  float mx = fmaxf(m[0], fmaxf(m[1], m[2]));
  float e0 = expf(m[0] - mx), e1 = expf(m[1] - mx), e2 = expf(m[2] - mx);
  float inv = 1.f / (e0 + e1 + e2);
  size_t o0 = ((size_t)(b * 3 + 0)) * N + n;
  w[o0] = e0 * inv; w[o0 + N] = e1 * inv; w[o0 + 2 * (size_t)N] = e2 * inv;
}

__global__ void k_gate_apply(const float* __restrict__ zrr, const float* __restrict__ zri,
                             const float* __restrict__ zfr, const float* __restrict__ zfi,
                             const float* __restrict__ zwr, const float* __restrict__ zwi,
                             const float* __restrict__ w, float* __restrict__ outR,
                             float* __restrict__ outI, int N, int total) {
  int i = blockIdx.x * 256 + threadIdx.x;
  if (i >= total) return;
  int n = i % N, t = i / N, b = t / 128;
  size_t wo = ((size_t)(b * 3)) * N + n;
  float w0 = w[wo], w1 = w[wo + N], w2 = w[wo + 2 * (size_t)N];
  outR[i] = w0 * zrr[i] + w1 * zfr[i] + w2 * zwr[i];
  outI[i] = w0 * zri[i] + w1 * zfi[i] + w2 * zwi[i];
}

__global__ __launch_bounds__(256)
void k_mean_spatial(const float* __restrict__ X, float* __restrict__ out, int N) {
  int bc = blockIdx.x, tid = threadIdx.x;
  const float* p = X + (size_t)bc * N;
  float s = 0.f;
  for (int i = tid; i < N; i += 256) s += p[i];
  __shared__ float sh[256];
  sh[tid] = s; __syncthreads();
  for (int st = 128; st > 0; st >>= 1) {
    if (tid < st) sh[tid] += sh[tid + st];
    __syncthreads();
  }
  if (tid == 0) out[bc] = sh[0] / N;
}

// tiny complex linear: out[b][oc] = sum_ic in[b][ic] * W[oc][ic] + bias   (+optional crelu)
__global__ void k_clinear(const float* __restrict__ pr, const float* __restrict__ pi,
                          const float* __restrict__ wr, const float* __restrict__ wi,
                          const float* __restrict__ br, const float* __restrict__ bi,
                          float* __restrict__ outR, float* __restrict__ outI,
                          int Cin, int Cout, int NB, int relu) {
  int t = blockIdx.x * 64 + threadIdx.x;
  if (t >= NB * Cout) return;
  int b = t / Cout, oc = t - b * Cout;
  float lr = br[oc], li = bi[oc];
  for (int c = 0; c < Cin; ++c) {
    float xr = pr[b * Cin + c], xi = pi[b * Cin + c];
    float ar = wr[oc * Cin + c], ai = wi[oc * Cin + c];
    lr += xr * ar - xi * ai;
    li += xr * ai + xi * ar;
  }
  if (relu) { lr = fmaxf(lr, 0.f); li = fmaxf(li, 0.f); }
  outR[t] = lr; outI[t] = li;
}

__global__ void k_cmag_sigmoid(const float* __restrict__ r, const float* __restrict__ i,
                               float* __restrict__ s, int n) {
  int t = blockIdx.x * 256 + threadIdx.x;
  if (t < n) {
    float m = sqrtf(r[t] * r[t] + i[t] * i[t]);
    s[t] = 1.f / (1.f + expf(-m));
  }
}

__global__ void k_scale_ca(float* __restrict__ R, float* __restrict__ I,
                           const float* __restrict__ s, int N, int total) {
  int i = blockIdx.x * 256 + threadIdx.x;
  if (i >= total) return;
  float f = 1.f + s[i / N];
  R[i] *= f; I[i] *= f;
}

__global__ void k_head(const float* __restrict__ pr, const float* __restrict__ pi,
                       const float* __restrict__ wr, const float* __restrict__ wi,
                       const float* __restrict__ br, const float* __restrict__ bi,
                       float* __restrict__ out, int C, int NB, int NO) {
  int t = blockIdx.x * 64 + threadIdx.x;
  if (t >= NB * NO) return;
  int b = t / NO, j = t - b * NO;
  float lr = br[j], li = bi[j];
  for (int c = 0; c < C; ++c) {
    float zr = pr[b * C + c], zi = pi[b * C + c];
    float ar = wr[j * C + c], ai = wi[j * C + c];
    lr += zr * ar - zi * ai;
    li += zr * ai + zi * ar;
  }
  out[t] = sqrtf(lr * lr + li * li);
}

// ---------------- host orchestration ----------------
struct CWp { const float *wr, *wi, *br, *bi; };
struct BNp { const float *gr, *br, *gi, *bi; };
struct RBp { CWp c1; BNp b1; CWp c2; BNp b2; };
struct ENCp { CWp stem; BNp stemb; CWp s1; BNp s1b; RBp rb1, rb2; CWp s2; BNp s2b; RBp rb3, rb4; };
struct ATTp { BNp bn; CWp q, k, v; const float* gamma; };

static inline dim3 g1d(long n) { return dim3((unsigned)((n + 255) / 256)); }

extern "C" void kernel_launch(void* const* d_in, const int* in_sizes, int n_in,
                              void* d_out, int out_size, void* d_ws, size_t ws_size,
                              hipStream_t stream) {
  if (n_in < 342) return;
  const int NB = 4, C = 128, Hf = 48, Wf = 48, N = Hf * Wf;  // final feature map
  const int TMAX = NB * 128 * N;                              // 1,179,648 floats

  // ---- parameter leaf cursor (setup_inputs insertion order) ----
  int pi = 3;
  auto P = [&]() -> const float* { return (const float*)d_in[pi++]; };
  auto getCW = [&]() { CWp c; c.wr = P(); c.wi = P(); c.br = P(); c.bi = P(); return c; };
  auto getBN = [&]() { BNp b; b.gr = P(); b.br = P(); b.gi = P(); b.bi = P(); return b; };
  auto getRB = [&]() { RBp r; r.c1 = getCW(); r.b1 = getBN(); r.c2 = getCW(); r.b2 = getBN(); return r; };
  auto getENC = [&]() {
    ENCp e; e.stem = getCW(); e.stemb = getBN(); e.s1 = getCW(); e.s1b = getBN();
    e.rb1 = getRB(); e.rb2 = getRB(); e.s2 = getCW(); e.s2b = getBN();
    e.rb3 = getRB(); e.rb4 = getRB(); return e;
  };
  auto getATT = [&]() { ATTp a; a.bn = getBN(); a.q = getCW(); a.k = getCW(); a.v = getCW(); a.gamma = P(); return a; };

  ENCp er = getENC(), ef = getENC(), ew = getENC();
  ATTp ar = getATT(), af = getATT(), aw = getATT();
  CWp g1 = getCW(); BNp gbn = getBN(); CWp g2 = getCW(); CWp ca1 = getCW(); CWp ca2 = getCW();
  const float* hwr = P(); const float* hwi = P(); const float* hbr = P(); const float* hbi = P();

  // ---- workspace bump allocator ----
  char* base = (char*)d_ws;
  size_t off = 0;
  auto alloc = [&](size_t nfloats) -> float* {
    float* p = (float*)(base + off);
    off += ((nfloats * sizeof(float)) + 255) & ~(size_t)255;
    return p;
  };
  float* t[8];  for (int i = 0; i < 8; ++i) t[i] = alloc(TMAX);
  float* zrR = alloc(TMAX); float* zrI = alloc(TMAX);
  float* zfR = alloc(TMAX); float* zfI = alloc(TMAX);
  float* zwR = alloc(TMAX); float* zwI = alloc(TMAX);
  float* xbnR = alloc(TMAX); float* xbnI = alloc(TMAX);
  float* qR = alloc(TMAX); float* qI = alloc(TMAX);
  float* kR = alloc(TMAX); float* kI = alloc(TMAX);
  float* vR = alloc(TMAX); float* vI = alloc(TMAX);
  float* Sbuf = alloc((size_t)N * N);
  float* catR = alloc((size_t)NB * 384 * N); float* catI = alloc((size_t)NB * 384 * N);
  float* gaR = alloc(TMAX); float* gaI = alloc(TMAX);
  float* g2R = alloc((size_t)NB * 3 * N); float* g2I = alloc((size_t)NB * 3 * N);
  float* gw  = alloc((size_t)NB * 3 * N);
  float* fusR = alloc(TMAX); float* fusI = alloc(TMAX);
  float* statM = alloc(512); float* statS = alloc(512);
  float* poolR = alloc(512); float* poolI = alloc(512);
  float* c1R = alloc(256); float* c1I = alloc(256);
  float* c2R = alloc(512); float* c2I = alloc(512);
  float* casig = alloc(512);
  float* hpR = alloc(512); float* hpI = alloc(512);
  (void)ws_size; (void)in_sizes;

  // ---- launch helpers ----
  auto runConv = [&](const float* Xr, const float* Xi, const CWp& w, float* Yr, float* Yi,
                     int nb, int Cin, int Hin, int Win, int Cout, int KS, int st, int pad,
                     int& Ho, int& Wo) {
    Ho = (Hin + 2 * pad - KS) / st + 1;
    Wo = (Win + 2 * pad - KS) / st + 1;
    int Ntot = nb * Ho * Wo;
    dim3 g((Ntot + 63) / 64, (Cout + 63) / 64);
    k_cconv_wmma<<<g, 256, 0, stream>>>(Xr, Xi, w.wr, w.wi, w.br, w.bi, Yr, Yi,
                                        nb, Cin, Hin, Win, Cout, Ho, Wo, KS, st, pad);
  };
  auto runBN = [&](float* Xr, float* Xi, float* Yr, float* Yi, const BNp& bn,
                   int nb, int Cc, int HW, int relu) {
    int total = nb * Cc * HW;
    k_bn_stats<<<Cc, 256, 0, stream>>>(Xr, statM, statS, nb, Cc, HW);
    k_bn_apply<<<g1d(total), 256, 0, stream>>>(Xr, Yr, statM, statS, bn.gr, bn.br, Cc, HW, total, relu);
    k_bn_stats<<<Cc, 256, 0, stream>>>(Xi, statM, statS, nb, Cc, HW);
    k_bn_apply<<<g1d(total), 256, 0, stream>>>(Xi, Yi, statM, statS, bn.gi, bn.bi, Cc, HW, total, relu);
  };
  auto runRes = [&](float* xr, float* xi, const RBp& rb, int Cc, int H, int W) {
    int Ho, Wo, HW = H * W, total = NB * Cc * HW;
    runConv(xr, xi, rb.c1, t[4], t[5], NB, Cc, H, W, Cc, 3, 1, 1, Ho, Wo);
    runBN(t[4], t[5], t[4], t[5], rb.b1, NB, Cc, HW, 1);
    runConv(t[4], t[5], rb.c2, t[6], t[7], NB, Cc, H, W, Cc, 3, 1, 1, Ho, Wo);
    runBN(t[6], t[7], t[6], t[7], rb.b2, NB, Cc, HW, 0);
    k_add_relu<<<g1d(total), 256, 0, stream>>>(xr, t[6], xr, total);
    k_add_relu<<<g1d(total), 256, 0, stream>>>(xi, t[7], xi, total);
  };
  auto runEncoder = [&](const float* xin, int Cin, int H, int W, int stemStride,
                        const ENCp& e, float* outR, float* outI) {
    int n0 = NB * Cin * H * W;
    k_deint<<<g1d(n0), 256, 0, stream>>>(xin, t[0], t[1], n0);
    int H1, W1, H2, W2, H3, W3;
    runConv(t[0], t[1], e.stem, t[2], t[3], NB, Cin, H, W, 32, 3, stemStride, 1, H1, W1);
    runBN(t[2], t[3], t[2], t[3], e.stemb, NB, 32, H1 * W1, 1);
    runConv(t[2], t[3], e.s1, t[0], t[1], NB, 32, H1, W1, 64, 3, 2, 1, H2, W2);
    runBN(t[0], t[1], t[0], t[1], e.s1b, NB, 64, H2 * W2, 1);
    runRes(t[0], t[1], e.rb1, 64, H2, W2);
    runRes(t[0], t[1], e.rb2, 64, H2, W2);
    runConv(t[0], t[1], e.s2, t[2], t[3], NB, 64, H2, W2, 128, 3, 1, 1, H3, W3);
    runBN(t[2], t[3], t[2], t[3], e.s2b, NB, 128, H3 * W3, 1);
    runRes(t[2], t[3], e.rb3, 128, H3, W3);
    runRes(t[2], t[3], e.rb4, 128, H3, W3);
    size_t bytes = sizeof(float) * (size_t)NB * 128 * H3 * W3;
    hipMemcpyAsync(outR, t[2], bytes, hipMemcpyDeviceToDevice, stream);
    hipMemcpyAsync(outI, t[3], bytes, hipMemcpyDeviceToDevice, stream);
  };
  auto runAttention = [&](float* xr, float* xi, const ATTp& a) {
    int Ho, Wo;
    runBN(xr, xi, xbnR, xbnI, a.bn, NB, C, N, 0);
    runConv(xbnR, xbnI, a.q, qR, qI, NB, C, Hf, Wf, C, 1, 1, 0, Ho, Wo);
    runConv(xbnR, xbnI, a.k, kR, kI, NB, C, Hf, Wf, C, 1, 1, 0, Ho, Wo);
    runConv(xbnR, xbnI, a.v, vR, vI, NB, C, Hf, Wf, C, 1, 1, 0, Ho, Wo);
    float scale = 1.f / sqrtf((float)C);
    for (int b = 0; b < NB; ++b) {
      size_t bo = (size_t)b * C * N;
      dim3 gs((N + 63) / 64, (N + 63) / 64);
      k_attn_scores<<<gs, 256, 0, stream>>>(qR + bo, qI + bo, kR + bo, kI + bo, Sbuf, C, N, scale);
      k_softmax_rows<<<N, 256, 0, stream>>>(Sbuf, N);
      dim3 go((N + 63) / 64, (C + 63) / 64);
      k_attn_out<<<go, 256, 0, stream>>>(vR + bo, vI + bo, Sbuf, xbnR + bo, xbnI + bo,
                                         a.gamma, xr + bo, xi + bo, C, N);
    }
  };

  // ---- full network ----
  runEncoder((const float*)d_in[0], 2, 192, 192, 2, er, zrR, zrI);
  runAttention(zrR, zrI, ar);
  runEncoder((const float*)d_in[1], 2, 192, 192, 2, ef, zfR, zfI);
  runAttention(zfR, zfI, af);
  runEncoder((const float*)d_in[2], 8, 96, 96, 1, ew, zwR, zwI);
  runAttention(zwR, zwI, aw);

  // fusion
  {
    int Ho, Wo;
    int tc = NB * 384 * N;
    k_concat3<<<g1d(tc), 256, 0, stream>>>(zrR, zfR, zwR, catR, N, tc);
    k_concat3<<<g1d(tc), 256, 0, stream>>>(zrI, zfI, zwI, catI, N, tc);
    runConv(catR, catI, g1, gaR, gaI, NB, 384, Hf, Wf, 128, 1, 1, 0, Ho, Wo);
    runBN(gaR, gaI, gaR, gaI, gbn, NB, 128, N, 1);
    runConv(gaR, gaI, g2, g2R, g2I, NB, 128, Hf, Wf, 3, 1, 1, 0, Ho, Wo);
    int tg = NB * N;
    k_gate_softmax<<<g1d(tg), 256, 0, stream>>>(g2R, g2I, gw, N, tg);
    int tf = NB * 128 * N;
    k_gate_apply<<<g1d(tf), 256, 0, stream>>>(zrR, zrI, zfR, zfI, zwR, zwI, gw, fusR, fusI, N, tf);
    // channel attention on pooled features
    k_mean_spatial<<<NB * C, 256, 0, stream>>>(fusR, poolR, N);
    k_mean_spatial<<<NB * C, 256, 0, stream>>>(fusI, poolI, N);
    k_clinear<<<(NB * 32 + 63) / 64, 64, 0, stream>>>(poolR, poolI, ca1.wr, ca1.wi, ca1.br, ca1.bi,
                                                      c1R, c1I, 128, 32, NB, 1);
    k_clinear<<<(NB * 128 + 63) / 64, 64, 0, stream>>>(c1R, c1I, ca2.wr, ca2.wi, ca2.br, ca2.bi,
                                                       c2R, c2I, 32, 128, NB, 0);
    k_cmag_sigmoid<<<g1d(NB * 128), 256, 0, stream>>>(c2R, c2I, casig, NB * 128);
    k_scale_ca<<<g1d(tf), 256, 0, stream>>>(fusR, fusI, casig, N, tf);
  }
  // head
  k_mean_spatial<<<NB * C, 256, 0, stream>>>(fusR, hpR, N);
  k_mean_spatial<<<NB * C, 256, 0, stream>>>(fusI, hpI, N);
  k_head<<<1, 64, 0, stream>>>(hpR, hpI, hwr, hwi, hbr, hbi, (float*)d_out, C, NB, 7);
  (void)out_size;
}
